// AttentionHead_34651796144482
// MI455X (gfx1250) — compile-verified, matrix operand order checked by reading, not run on hardware
//
#include <hip/hip_runtime.h>

typedef __attribute__((ext_vector_type(16))) __bf16 v16bf;
typedef __attribute__((ext_vector_type(2)))  __bf16 bf16x2;
typedef __attribute__((ext_vector_type(8)))  float  v8f;

#define S_TOT   8192
#define D_E     256
#define N_DTILE (D_E / 16)     // 16
#define N_STILE (S_TOT / 16)   // 512
#define KB      32             // keys per block-cooperative tile
#define SCALE_Q (1.0f / 16.0f) // 1/sqrt(dim), dim = 256
#define NEG_BIG (-3.0e38f)

// K-index base for 16-bit A/B operand VGPR v (0..7), half-wave h (0/1).
// ISA 16-bit A 16x32 layout: VGPR0-3 lanes0-15 hold K=0..7, lanes16-31 K=8..15;
// VGPR4-7 hold K=16..23 / 24..31. Two packed bf16 per VGPR.
__device__ __forceinline__ int kbase(int v, int h) {
    return ((v >> 2) << 4) + (h << 3) + ((v & 3) << 1);
}

__device__ __forceinline__ v8f wmma_bf16(v16bf a, v16bf b, v8f c) {
    return __builtin_amdgcn_wmma_f32_16x16x32_bf16(
        /*neg_a=*/false, a, /*neg_b=*/false, b,
        /*c_mod=*/(short)0, c, /*reuse_a=*/false, /*reuse_b=*/false);
}

// ASYNCcnt-tracked 16-byte global -> LDS copy (CDNA5 async path).
__device__ __forceinline__ void async_copy16(const __bf16* g, const __bf16* lds_dst) {
    const uint32_t lds_off = (uint32_t)(uintptr_t)lds_dst; // flat LDS addr = addr[31:0]
    asm volatile("global_load_async_to_lds_b128 %0, %1, off"
                 :: "v"(lds_off), "v"((unsigned long long)(uintptr_t)g)
                 : "memory");
}

__device__ __forceinline__ void wait_async0() {
    asm volatile("s_wait_asynccnt 0x0" ::: "memory");
}

// ---------------------------------------------------------------------------
// Phase 1: Q = (x @ Wq^T) * 1/16, K = x @ Wk^T, V = x @ Wv^T (stored transposed)
// all outputs bf16. One wave per 16x16 tile, 8x WMMA over K=256.
// ---------------------------------------------------------------------------
__global__ __launch_bounds__(128)
void proj_qkv_kernel(const float* __restrict__ x,
                     const float* __restrict__ Wq,
                     const float* __restrict__ Wk,
                     const float* __restrict__ Wv,
                     __bf16* __restrict__ Qb,   // [S, 256]
                     __bf16* __restrict__ Kb,   // [S, 256]
                     __bf16* __restrict__ Vt)   // [256, S] (transposed)
{
    const int lane = threadIdx.x & 31;
    const int wid  = threadIdx.x >> 5;
    const int tile = blockIdx.x * 4 + wid;
    const int mat  = tile / (N_STILE * N_DTILE);
    const int rem  = tile % (N_STILE * N_DTILE);
    const int trow = rem / N_DTILE;   // tile along S
    const int tcol = rem % N_DTILE;   // tile along dim
    const int h = lane >> 4, m = lane & 15;

    const float* W = (mat == 0) ? Wq : (mat == 1) ? Wk : Wv;

    const int arow = trow * 16 + m;   // A operand: row per lane
    const int bcol = tcol * 16 + m;   // B operand: output column per lane

    v8f c = {};
    #pragma unroll
    for (int chunk = 0; chunk < 8; ++chunk) {
        v16bf a, b;
        #pragma unroll
        for (int v = 0; v < 8; ++v) {
            const int kb = chunk * 32 + kbase(v, h);
            a[2 * v]     = (__bf16)x[arow * D_E + kb];
            a[2 * v + 1] = (__bf16)x[arow * D_E + kb + 1];
            b[2 * v]     = (__bf16)W[bcol * D_E + kb];
            b[2 * v + 1] = (__bf16)W[bcol * D_E + kb + 1];
        }
        c = wmma_bf16(a, b, c);
    }

    // C layout: VGPR r -> row r + 8h, col = m
    #pragma unroll
    for (int r = 0; r < 8; ++r) {
        const int srow = trow * 16 + r + 8 * h;
        const int dcol = tcol * 16 + m;
        const float val = c[r];
        if (mat == 0)      Qb[srow * D_E + dcol]   = (__bf16)(val * SCALE_Q);
        else if (mat == 1) Kb[srow * D_E + dcol]   = (__bf16)val;
        else               Vt[dcol * S_TOT + srow] = (__bf16)val;
    }
}

// ---------------------------------------------------------------------------
// Phase 2: causal flash attention.
// Block = 4 waves = 64 query rows; O (16x256 fp32) and Q (16x256 bf16,
// A-layout) live in VGPRs per wave. The block cooperatively double-buffers
// 32-key K/V tiles into LDS with async loads, overlapping fetch of tile j+1
// with compute of tile j.
// ---------------------------------------------------------------------------
__global__ __launch_bounds__(128)
void flash_attn_kernel(const __bf16* __restrict__ Qb,
                       const __bf16* __restrict__ Kb,
                       const __bf16* __restrict__ Vt,
                       float* __restrict__ out)
{
    __shared__ __bf16 sK[2][KB * D_E];   // 2 x 16 KB, row-major [key][de]
    __shared__ __bf16 sV[2][D_E * KB];   // 2 x 16 KB, row-major [de][key]
    __shared__ __bf16 sP[4][16 * KB];    // per-wave P staging, 4 KB

    const int tid  = threadIdx.x;
    const int lane = tid & 31;
    const int wid  = tid >> 5;
    const int h = lane >> 4, m = lane & 15;
    const int q0   = blockIdx.x * 64 + wid * 16;   // this wave's query rows
    const int jmax = 2 * blockIdx.x + 1;           // last key tile for the block

    // Preload Q rows (A-operand layout), 8 chunks of K=32 over the 256 dim.
    v16bf qa[8];
    #pragma unroll
    for (int chunk = 0; chunk < 8; ++chunk) {
        #pragma unroll
        for (int v = 0; v < 8; ++v) {
            const int kb = chunk * 32 + kbase(v, h);
            const bf16x2 p = *(const bf16x2*)(Qb + (q0 + m) * D_E + kb);
            qa[chunk][2 * v]     = p[0];
            qa[chunk][2 * v + 1] = p[1];
        }
    }

    v8f o[16] = {};                 // 16 rows x 256 cols fp32 accumulator
    float mrow[8], lrow[8];
    #pragma unroll
    for (int r = 0; r < 8; ++r) { mrow[r] = NEG_BIG; lrow[r] = 0.0f; }

    // ---- stage key tile j into LDS (cooperative, async). 16 KB each for K,V.
    auto stage_kv = [&](__bf16* dK, __bf16* dV, int k0) {
        const __bf16* gk = Kb + (size_t)k0 * D_E;     // contiguous 16 KB
        #pragma unroll
        for (int i = 0; i < 8; ++i) {
            const int e = i * 128 + tid;              // 16-byte unit index
            async_copy16(gk + e * 8, dK + e * 8);
        }
        #pragma unroll
        for (int i = 0; i < 8; ++i) {
            const int e  = i * 128 + tid;
            const int rr = e >> 2, seg = e & 3;       // 4 x 16B per de-row
            async_copy16(Vt + (size_t)rr * S_TOT + k0 + seg * 8,
                         dV + rr * KB + seg * 8);
        }
    };

    stage_kv(sK[0], sV[0], 0);

    for (int j = 0; j <= jmax; ++j) {
        const int cur = j & 1, nxt = cur ^ 1;
        const int k0  = j * KB;

        wait_async0();         // tile j resident (this wave's share)
        __syncthreads();       // whole tile resident; prior readers of nxt done

        if (j < jmax) stage_kv(sK[nxt], sV[nxt], k0 + KB);  // overlap fetch j+1

        if (k0 <= q0 + 15) {   // this wave still below/at its diagonal
            // ---- S = Q K^T for 32 keys: two 16x16 tiles, K=256 in 8 chunks
            v8f s0 = {}, s1 = {};
            const __bf16* kb_lds = sK[cur];
            #pragma unroll
            for (int chunk = 0; chunk < 8; ++chunk) {
                v16bf b0, b1;
                #pragma unroll
                for (int v = 0; v < 8; ++v) {
                    const int kb = chunk * 32 + kbase(v, h);
                    const bf16x2 p0 = *(const bf16x2*)(kb_lds + m * D_E + kb);
                    const bf16x2 p1 = *(const bf16x2*)(kb_lds + (16 + m) * D_E + kb);
                    b0[2 * v] = p0[0]; b0[2 * v + 1] = p0[1];
                    b1[2 * v] = p1[0]; b1[2 * v + 1] = p1[1];
                }
                s0 = wmma_bf16(qa[chunk], b0, s0);
                s1 = wmma_bf16(qa[chunk], b1, s1);
            }

            // ---- causal mask (only near the diagonal) + online softmax
            const bool needMask = (k0 + KB - 1 > q0);
            float sc[8];
            const int col0 = k0 + m, col1 = k0 + 16 + m;
            #pragma unroll
            for (int r = 0; r < 8; ++r) {
                const int row = q0 + r + 8 * h;
                float x0 = s0[r], x1 = s1[r];
                if (needMask) {
                    x0 = (col0 <= row) ? x0 : NEG_BIG;
                    x1 = (col1 <= row) ? x1 : NEG_BIG;
                }
                float mx = fmaxf(x0, x1);
                mx = fmaxf(mx, __shfl_xor(mx, 1, 32));
                mx = fmaxf(mx, __shfl_xor(mx, 2, 32));
                mx = fmaxf(mx, __shfl_xor(mx, 4, 32));
                mx = fmaxf(mx, __shfl_xor(mx, 8, 32));
                const float mnew = fmaxf(mrow[r], mx);
                const float p0 = __expf(x0 - mnew);
                const float p1 = __expf(x1 - mnew);
                float ps = p0 + p1;
                ps += __shfl_xor(ps, 1, 32);
                ps += __shfl_xor(ps, 2, 32);
                ps += __shfl_xor(ps, 4, 32);
                ps += __shfl_xor(ps, 8, 32);
                const float rescale = __expf(mrow[r] - mnew);
                lrow[r] = lrow[r] * rescale + ps;
                mrow[r] = mnew;
                sc[r] = rescale;
                // stage P (bf16) for layout change C -> A
                sP[wid][(r + 8 * h) * KB + m]      = (__bf16)p0;
                sP[wid][(r + 8 * h) * KB + 16 + m] = (__bf16)p1;
            }

            // ---- rescale running O
            #pragma unroll
            for (int t = 0; t < 16; ++t) {
                #pragma unroll
                for (int r = 0; r < 8; ++r) o[t][r] *= sc[r];
            }

            asm volatile("" ::: "memory"); // LDS writes before reads (same wave, DS in-order)

            // ---- read P back as A operand (16 rows x 32 keys)
            v16bf pa;
            #pragma unroll
            for (int v = 0; v < 8; ++v) {
                const int kb = kbase(v, h);
                const bf16x2 p = *(const bf16x2*)(&sP[wid][m * KB + kb]);
                pa[2 * v] = p[0]; pa[2 * v + 1] = p[1];
            }

            // ---- O += P @ V^T : 16 output column tiles, K = 32 keys
            #pragma unroll
            for (int t = 0; t < 16; ++t) {
                v16bf bv;
                #pragma unroll
                for (int v = 0; v < 8; ++v) {
                    const int kb = kbase(v, h);
                    const bf16x2 p = *(const bf16x2*)(sV[cur] + (t * 16 + m) * KB + kb);
                    bv[2 * v] = p[0]; bv[2 * v + 1] = p[1];
                }
                o[t] = wmma_bf16(pa, bv, o[t]);
            }
        }

        __syncthreads();       // all reads of tile j done before it is overwritten
    }

    // ---- normalize and write out (fp32)
    #pragma unroll
    for (int r = 0; r < 8; ++r) {
        const float inv = 1.0f / lrow[r];
        const int row = q0 + r + 8 * h;
        #pragma unroll
        for (int t = 0; t < 16; ++t)
            out[row * D_E + t * 16 + m] = o[t][r] * inv;
    }
}

// ---------------------------------------------------------------------------
extern "C" void kernel_launch(void* const* d_in, const int* in_sizes, int n_in,
                              void* d_out, int out_size, void* d_ws, size_t ws_size,
                              hipStream_t stream) {
    const float* x  = (const float*)d_in[0];
    const float* Wq = (const float*)d_in[1];
    const float* Wk = (const float*)d_in[2];
    const float* Wv = (const float*)d_in[3];

    __bf16* Qb = (__bf16*)d_ws;                      // 4 MB
    __bf16* Kb = Qb + (size_t)S_TOT * D_E;           // 4 MB
    __bf16* Vt = Kb + (size_t)S_TOT * D_E;           // 4 MB (transposed V)
    float*  outp = (float*)d_out;

    proj_qkv_kernel<<<3 * N_STILE * N_DTILE / 4, 128, 0, stream>>>(
        x, Wq, Wk, Wv, Qb, Kb, Vt);
    flash_attn_kernel<<<S_TOT / 64, 128, 0, stream>>>(Qb, Kb, Vt, outp);
}
